// RWKV7Block_78391743086640
// MI455X (gfx1250) — matrix operand order repaired and checked by reading, not compiled, not run on hardware
//
#include <hip/hip_runtime.h>
#include <hip/hip_bf16.h>
#include <math.h>
#include <stdint.h>

// ---------------- problem constants (from reference) ----------------
#define Bc   8
#define Tc   512
#define Cc   1024
#define Hc   16
#define Nn   64
#define BT   (Bc*Tc)                 // 4096 rows
#define BTC  ((size_t)BT*(size_t)Cc) // 4,194,304 elements
#define DWd  64
#define DAd  64
#define DGd  128
#define C4   (4*Cc)                  // 4096

typedef __bf16 bf16;
typedef __attribute__((ext_vector_type(16))) __bf16 v16bf;
typedef __attribute__((ext_vector_type(8)))  __bf16 v8bf;
typedef __attribute__((ext_vector_type(4)))  __bf16 v4bf;
typedef __attribute__((ext_vector_type(8)))  float  v8f;
typedef __attribute__((ext_vector_type(4)))  float  v4f;
typedef __attribute__((ext_vector_type(4)))  unsigned int u32x4;
typedef __attribute__((ext_vector_type(8)))  int    i32x8;
typedef __attribute__((ext_vector_type(4)))  int    i32x4;

// epilogue modes for the WMMA GEMM
#define EP_F32        0
#define EP_BF16       1
#define EP_TANH_BF16  2
#define EP_SIG_BF16   3
#define EP_RELU2_BF16 4

// =====================================================================
// WMMA bf16 GEMM:  out[M,Nc] = A[M,K] (bf16 row-major) x W[K,Nc] (bf16 row-major)
// Block tile 128(M) x 64(N), 8 waves as 4x2, each wave a 32x32 subtile
// (4 x v_wmma_f32_16x16x32_bf16 per K-step). K step = 32.
// A tile is DMA'd by the Tensor Data Mover (tensor_load_to_lds) with LDS
// padding 16B per 64B row -> 40-element LDS row stride (conflict-free).
// B tile is staged transposed (b128 load + 16b scatter).
// M%128==0, K%32==0, Nc%64==0 hold for every matmul in this block.
// =====================================================================
#define LDSP 40   // LDS row stride (elements): 80B, 16B-aligned, conflict-free

template<int EP>
__global__ __launch_bounds__(256)
void wmma_gemm_kernel(const bf16* __restrict__ A, const bf16* __restrict__ W,
                      void* __restrict__ outp, int M, int K, int Nc)
{
    __shared__ __align__(16) bf16 As[128][LDSP];   // A tile (TDM dest), row-major
    __shared__ __align__(16) bf16 BsT[64][LDSP];   // W tile, TRANSPOSED [n][k]

    const int tid  = threadIdx.x;
    const int wave = tid >> 5;
    const int lane = tid & 31;
    const int wm   = wave >> 1;          // 0..3  -> 32 rows each
    const int wn   = wave & 1;           // 0..1  -> 32 cols each
    const int rowBase = blockIdx.y * 128;
    const int colBase = blockIdx.x * 64;

    const int lm  = lane & 15;
    const int hiA = (lane >> 4) << 3;    // 0 or 8   (A K-offset per lane half)
    const int hiB = (lane >> 4) << 4;    // 0 or 16  (unused; B fragments also use hiA)

    // B staging coordinates (constant per thread)
    const int br = tid >> 3;             // W k-row 0..31
    const int bc = (tid & 7) << 3;       // W col chunk 0..56

    // ---- TDM descriptor (loop-invariant parts) ----
    const unsigned ldsAddr = (unsigned)(uintptr_t)&As[0][0];
    const unsigned long long gArow =
        (unsigned long long)(uintptr_t)A + (unsigned long long)rowBase * (unsigned)K * 2ull;
    // group1: data_size=2B, pad_enable, pad_interval=3 (64B), pad_amount=3 (16B)
    i32x8 g1;
    g1[0] = (1 << 16) | (1 << 20) | (3 << 22) | (3 << 25);
    g1[1] = (int)(((unsigned)K & 0xFFFFu) << 16);                     // tensor_dim0 lo16
    g1[2] = (int)(((unsigned)K >> 16) | (((unsigned)M & 0xFFFFu) << 16)); // tdim0 hi | tdim1 lo
    g1[3] = (int)(((unsigned)M >> 16) | (32u << 16));                 // tdim1 hi | tile_dim0=32
    g1[4] = 128;                                                      // tile_dim1=128, tile_dim2=0
    g1[5] = K;                                                        // tensor_dim0_stride lo32
    g1[6] = 0;
    g1[7] = 0;
    const i32x4 gz  = {0, 0, 0, 0};
    const i32x8 gz8 = {0, 0, 0, 0, 0, 0, 0, 0};

    v8f acc[2][2] = {{{}, {}}, {{}, {}}};

    for (int kb = 0; kb < K; kb += 32) {
        // ---- A tile via Tensor Data Mover (one wave issues the DMA) ----
        if (wave == 0) {
            unsigned long long ga = gArow + (unsigned)kb * 2u;
            u32x4 g0;
            g0[0] = 1u;                                   // count=1 (valid descriptor)
            g0[1] = ldsAddr;                              // LDS dest (bytes)
            g0[2] = (unsigned)ga;                         // global_addr lo
            g0[3] = (unsigned)(ga >> 32) | (2u << 30);    // global_addr hi | type=2
            __builtin_amdgcn_tensor_load_to_lds(g0, g1, gz, gz, gz8, 0);
        }
        // ---- stage B transposed: 32x64, b128 load + 8x16b scatter ----
        {
            v8bf w8 = *(const v8bf*)&W[(size_t)(kb + br) * Nc + (colBase + bc)];
#pragma unroll
            for (int j = 0; j < 8; ++j) BsT[bc + j][br] = w8[j];
        }
        if (kb + 32 < K) {   // hint next W tile into cache (global_prefetch_b8)
            __builtin_prefetch(&W[(size_t)(kb + 32 + br) * Nc + (colBase + bc)], 0, 1);
        }
        __builtin_amdgcn_s_wait_tensorcnt(0);   // no-op for non-issuing waves
        __syncthreads();

        // ---- fragments: contiguous b128 LDS loads ----
        v16bf aF[2], bF[2];
#pragma unroll
        for (int mi = 0; mi < 2; ++mi) {
            const int mr = wm * 32 + mi * 16 + lm;
            *(v8bf*)&aF[mi]       = *(const v8bf*)&As[mr][hiA];
            *((v8bf*)&aF[mi] + 1) = *(const v8bf*)&As[mr][16 + hiA];
        }
#pragma unroll
        for (int ni = 0; ni < 2; ++ni) {
            const int nc = wn * 32 + ni * 16 + lm;
            *(v8bf*)&bF[ni]       = *(const v8bf*)&BsT[nc][hiA];
            *((v8bf*)&bF[ni] + 1) = *(const v8bf*)&BsT[nc][16 + hiA];
        }
#pragma unroll
        for (int mi = 0; mi < 2; ++mi)
#pragma unroll
            for (int ni = 0; ni < 2; ++ni)
                acc[mi][ni] = __builtin_amdgcn_wmma_f32_16x16x32_bf16(
                    false, aF[mi], false, bF[ni], (short)0, acc[mi][ni],
                    false, false);
        __syncthreads();   // also orders next iteration's TDM/stores after reads
    }

    // C/D layout: VGPR r -> M=r (lanes 0-15) / M=r+8 (lanes 16-31); N = lane&15
    const int rowOff = (lane >> 4) << 3;
#pragma unroll
    for (int mi = 0; mi < 2; ++mi) {
#pragma unroll
        for (int ni = 0; ni < 2; ++ni) {
            const int col = colBase + wn * 32 + ni * 16 + lm;
#pragma unroll
            for (int r = 0; r < 8; ++r) {
                int row  = rowBase + wm * 32 + mi * 16 + rowOff + r;
                size_t o = (size_t)row * Nc + col;
                float x  = acc[mi][ni][r];
                if (EP == EP_F32) {
                    ((float*)outp)[o] = x;
                } else {
                    float y;
                    if      (EP == EP_BF16)      y = x;
                    else if (EP == EP_TANH_BF16) y = tanhf(x);
                    else if (EP == EP_SIG_BF16)  y = 1.0f / (1.0f + expf(-x));
                    else                         y = (x > 0.0f) ? x * x : 0.0f;
                    ((bf16*)outp)[o] = (bf16)y;
                }
            }
        }
    }
}

// =====================================================================
// LayerNorm over C=1024, one block per (b,t) row
// =====================================================================
__global__ __launch_bounds__(256)
void ln_kernel(const float* __restrict__ in, const float* __restrict__ w,
               const float* __restrict__ b, float* __restrict__ out, float eps)
{
    const int row = blockIdx.x;
    const int tid = threadIdx.x;
    const float* xr = in + (size_t)row * Cc;
    __shared__ float red[256];

    v4f xv = ((const v4f*)xr)[tid];      // 256 threads x 4 = 1024
    float s = xv[0] + xv[1] + xv[2] + xv[3];
    red[tid] = s; __syncthreads();
    for (int st = 128; st > 0; st >>= 1) { if (tid < st) red[tid] += red[tid + st]; __syncthreads(); }
    float mean = red[0] * (1.0f / Cc);
    __syncthreads();

    float v = 0.f;
#pragma unroll
    for (int j = 0; j < 4; ++j) { float d = xv[j] - mean; v += d * d; }
    red[tid] = v; __syncthreads();
    for (int st = 128; st > 0; st >>= 1) { if (tid < st) red[tid] += red[tid + st]; __syncthreads(); }
    float rstd = rsqrtf(red[0] * (1.0f / Cc) + eps);

    v4f wv = ((const v4f*)w)[tid];
    v4f bv = ((const v4f*)b)[tid];
    v4f ov;
#pragma unroll
    for (int j = 0; j < 4; ++j) ov[j] = (xv[j] - mean) * rstd * wv[j] + bv[j];
    ((v4f*)(out + (size_t)row * Cc))[tid] = ov;
}

// =====================================================================
// time-shift + 6-way mix -> bf16 GEMM operands (4 elems/thread, vector I/O)
// =====================================================================
__global__ __launch_bounds__(256)
void shift_mix6_kernel(const float* __restrict__ xn,
                       const float* __restrict__ mr, const float* __restrict__ mw,
                       const float* __restrict__ mk, const float* __restrict__ mv,
                       const float* __restrict__ ma, const float* __restrict__ mg,
                       bf16* __restrict__ or_, bf16* __restrict__ ow,
                       bf16* __restrict__ ok,  bf16* __restrict__ ov,
                       bf16* __restrict__ oa,  bf16* __restrict__ og)
{
    const size_t n4 = BTC / 4;
    for (size_t i4 = (size_t)blockIdx.x * blockDim.x + threadIdx.x; i4 < n4;
         i4 += (size_t)gridDim.x * blockDim.x) {
        size_t i   = i4 * 4;
        size_t row = i / Cc;
        int    c   = (int)(i % Cc);
        int    t   = (int)(row % Tc);
        v4f xv = ((const v4f*)xn)[i4];
        v4f pv = {};
        if (t != 0) pv = ((const v4f*)xn)[i4 - Cc / 4];
        v4bf vr, vw, vk, vv, va, vg;
#pragma unroll
        for (int j = 0; j < 4; ++j) {
            float dx = pv[j] - xv[j];
            vr[j] = (bf16)(xv[j] + dx * mr[c + j]);
            vw[j] = (bf16)(xv[j] + dx * mw[c + j]);
            vk[j] = (bf16)(xv[j] + dx * mk[c + j]);
            vv[j] = (bf16)(xv[j] + dx * mv[c + j]);
            va[j] = (bf16)(xv[j] + dx * ma[c + j]);
            vg[j] = (bf16)(xv[j] + dx * mg[c + j]);
        }
        ((v4bf*)or_)[i4] = vr; ((v4bf*)ow)[i4] = vw; ((v4bf*)ok)[i4] = vk;
        ((v4bf*)ov)[i4]  = vv; ((v4bf*)oa)[i4] = va; ((v4bf*)og)[i4] = vg;
    }
}

__global__ __launch_bounds__(256)
void ffn_mix_kernel(const float* __restrict__ xf, const float* __restrict__ mk,
                    bf16* __restrict__ out)
{
    const size_t n4 = BTC / 4;
    for (size_t i4 = (size_t)blockIdx.x * blockDim.x + threadIdx.x; i4 < n4;
         i4 += (size_t)gridDim.x * blockDim.x) {
        size_t i   = i4 * 4;
        size_t row = i / Cc;
        int    c   = (int)(i % Cc);
        int    t   = (int)(row % Tc);
        v4f xv = ((const v4f*)xf)[i4];
        v4f pv = {};
        if (t != 0) pv = ((const v4f*)xf)[i4 - Cc / 4];
        v4bf o;
#pragma unroll
        for (int j = 0; j < 4; ++j)
            o[j] = (bf16)(xv[j] + (pv[j] - xv[j]) * mk[c + j]);
        ((v4bf*)out)[i4] = o;
    }
}

// decay = exp(-exp(-softplus(-(w0+y)) - 0.5)), in place on y
__global__ __launch_bounds__(256)
void decay_kernel(float* __restrict__ y, const float* __restrict__ w0)
{
    for (size_t i = (size_t)blockIdx.x * blockDim.x + threadIdx.x; i < BTC;
         i += (size_t)gridDim.x * blockDim.x) {
        int c = (int)(i % Cc);
        float z    = w0[c] + y[i];
        float wlog = -log1pf(expf(-z)) - 0.5f;   // -softplus(-z) - 0.5
        y[i] = expf(-expf(wlog));
    }
}

// a = sigmoid(a0 + y), in place
__global__ __launch_bounds__(256)
void sigmoid_bias_kernel(float* __restrict__ y, const float* __restrict__ a0)
{
    for (size_t i = (size_t)blockIdx.x * blockDim.x + threadIdx.x; i < BTC;
         i += (size_t)gridDim.x * blockDim.x) {
        int c = (int)(i % Cc);
        y[i] = 1.0f / (1.0f + expf(-(a0[c] + y[i])));
    }
}

__global__ __launch_bounds__(256)
void add_kernel(const float* __restrict__ a, const float* __restrict__ b,
                float* __restrict__ out)
{
    const size_t n4 = BTC / 4;
    for (size_t i4 = (size_t)blockIdx.x * blockDim.x + threadIdx.x; i4 < n4;
         i4 += (size_t)gridDim.x * blockDim.x) {
        v4f av = ((const v4f*)a)[i4];
        v4f bv = ((const v4f*)b)[i4];
        v4f o;
#pragma unroll
        for (int j = 0; j < 4; ++j) o[j] = av[j] + bv[j];
        ((v4f*)out)[i4] = o;
    }
}

__global__ __launch_bounds__(256)
void cvt_bf16_kernel(const float* __restrict__ in, bf16* __restrict__ out, size_t n)
{
    const size_t n4 = n / 4;
    for (size_t i4 = (size_t)blockIdx.x * blockDim.x + threadIdx.x; i4 < n4;
         i4 += (size_t)gridDim.x * blockDim.x) {
        v4f xv = ((const v4f*)in)[i4];
        v4bf o;
#pragma unroll
        for (int j = 0; j < 4; ++j) o[j] = (bf16)xv[j];
        ((v4bf*)out)[i4] = o;
    }
}

// =====================================================================
// per-(b,t,h): kk = normalize(k*k_k); k *= 1+(a-1)*k_a; emit -kk and kk*a
// =====================================================================
__global__ __launch_bounds__(64)
void prep_wkv_kernel(float* __restrict__ k, const float* __restrict__ a,
                     const float* __restrict__ k_k, const float* __restrict__ k_a,
                     float* __restrict__ neg_kk, float* __restrict__ kk_a)
{
    const int hb = blockIdx.x;            // b*T*H + t*H + h
    const int h  = hb & (Hc - 1);
    const int n  = threadIdx.x;
    const size_t g = (size_t)hb * Nn + n;
    const int c  = h * Nn + n;

    float kr    = k[g];
    float kkraw = kr * k_k[c];

    __shared__ float red[Nn];
    red[n] = kkraw * kkraw; __syncthreads();
#pragma unroll
    for (int s = 32; s > 0; s >>= 1) { if (n < s) red[n] += red[n + s]; __syncthreads(); }
    float nrm = sqrtf(red[0]);
    float kk  = kkraw / fmaxf(nrm, 1e-12f);

    float av = a[g];
    k[g]      = kr * (1.0f + (av - 1.0f) * k_a[c]);
    neg_kk[g] = -kk;
    kk_a[g]   = kk * av;
}

// =====================================================================
// WKV7 scan: one 64-thread block per (b,h); thread i owns state row i
// S = S*diag(d) + (S.a) b^T + v k^T ;  o = S r
// =====================================================================
__global__ __launch_bounds__(64)
void wkv7_kernel(const float* __restrict__ r, const float* __restrict__ d,
                 const float* __restrict__ k, const float* __restrict__ v,
                 const float* __restrict__ akk, const float* __restrict__ bkk,
                 float* __restrict__ o)
{
    const int bh = blockIdx.x;            // b*H + h
    const int b  = bh >> 4, h = bh & (Hc - 1);
    const int i  = threadIdx.x;

    float S[Nn];
#pragma unroll
    for (int j = 0; j < Nn; ++j) S[j] = 0.0f;

    __shared__ float rs[Nn], ds[Nn], ks[Nn], vs[Nn], as_[Nn], bs[Nn];

    for (int t = 0; t < Tc; ++t) {
        size_t base = (((size_t)b * Tc + t) * Hc + h) * Nn;
        rs[i]  = r[base + i];  ds[i] = d[base + i];  ks[i] = k[base + i];
        vs[i]  = v[base + i];  as_[i] = akk[base + i]; bs[i] = bkk[base + i];
        __syncthreads();

        float sa = 0.0f;
#pragma unroll
        for (int j = 0; j < Nn; ++j) sa += S[j] * as_[j];

        float vi = vs[i];
        float outv = 0.0f;
#pragma unroll
        for (int j = 0; j < Nn; ++j) {
            float s = S[j] * ds[j] + sa * bs[j] + vi * ks[j];
            S[j] = s;
            outv += s * rs[j];
        }
        o[base + i] = outv;
        __syncthreads();
    }
}

// =====================================================================
// per-(b,t,h): GroupNorm (eps=64e-5) + (r.k.r_k bonus)*v, gate by g -> bf16
// =====================================================================
__global__ __launch_bounds__(64)
void postmix_kernel(const float* __restrict__ o, const float* __restrict__ r,
                    const float* __restrict__ k, const float* __restrict__ v,
                    const float* __restrict__ r_k, const float* __restrict__ gn_w,
                    const float* __restrict__ gn_b, const float* __restrict__ g,
                    bf16* __restrict__ gated)
{
    const int hb = blockIdx.x;
    const int h  = hb & (Hc - 1);
    const int n  = threadIdx.x;
    const size_t gi = (size_t)hb * Nn + n;
    const int c  = h * Nn + n;

    __shared__ float red[Nn];
    float ov = o[gi];

    red[n] = ov; __syncthreads();
#pragma unroll
    for (int s = 32; s > 0; s >>= 1) { if (n < s) red[n] += red[n + s]; __syncthreads(); }
    float mean = red[0] * (1.0f / Nn);
    __syncthreads();

    float dm = ov - mean;
    red[n] = dm * dm; __syncthreads();
#pragma unroll
    for (int s = 32; s > 0; s >>= 1) { if (n < s) red[n] += red[n + s]; __syncthreads(); }
    float rstd = rsqrtf(red[0] * (1.0f / Nn) + 64e-5f);
    __syncthreads();

    red[n] = r[gi] * k[gi] * r_k[c]; __syncthreads();
#pragma unroll
    for (int s = 32; s > 0; s >>= 1) { if (n < s) red[n] += red[n + s]; __syncthreads(); }
    float dot = red[0];

    float val = dm * rstd * gn_w[c] + gn_b[c] + dot * v[gi];
    gated[gi] = (bf16)(val * g[gi]);
}

// =====================================================================
// host-side orchestration
// =====================================================================
template<int EP>
static void launch_gemm(const bf16* A, const bf16* W, void* out,
                        int M, int K, int Nc, hipStream_t s)
{
    dim3 grid(Nc / 64, M / 128), blk(256);
    wmma_gemm_kernel<EP><<<grid, blk, 0, s>>>(A, W, out, M, K, Nc);
}

extern "C" void kernel_launch(void* const* d_in, const int* in_sizes, int n_in,
                              void* d_out, int out_size, void* d_ws, size_t ws_size,
                              hipStream_t stream)
{
    (void)in_sizes; (void)n_in; (void)out_size; (void)ws_size;

    const float* x        = (const float*)d_in[0];
    // d_in[1] = v_first input (unused: layer 0 overwrites it)
    const float* ln_pre_w = (const float*)d_in[2];
    const float* ln_pre_b = (const float*)d_in[3];
    const float* ln_at_w  = (const float*)d_in[4];
    const float* ln_at_b  = (const float*)d_in[5];
    const float* ln_ff_w  = (const float*)d_in[6];
    const float* ln_ff_b  = (const float*)d_in[7];
    const float* x_r      = (const float*)d_in[8];
    const float* x_w      = (const float*)d_in[9];
    const float* x_k      = (const float*)d_in[10];
    const float* x_v      = (const float*)d_in[11];
    const float* x_a      = (const float*)d_in[12];
    const float* x_g      = (const float*)d_in[13];
    const float* w0       = (const float*)d_in[14];
    const float* w1       = (const float*)d_in[15];
    const float* w2       = (const float*)d_in[16];
    const float* a0       = (const float*)d_in[17];
    const float* a1       = (const float*)d_in[18];
    const float* a2       = (const float*)d_in[19];
    const float* g1       = (const float*)d_in[20];
    const float* g2       = (const float*)d_in[21];
    const float* k_k      = (const float*)d_in[22];
    const float* k_a      = (const float*)d_in[23];
    const float* r_k      = (const float*)d_in[24];
    const float* W_r      = (const float*)d_in[25];
    const float* W_k      = (const float*)d_in[26];
    const float* W_v      = (const float*)d_in[27];
    const float* W_o      = (const float*)d_in[28];
    const float* gn_w     = (const float*)d_in[29];
    const float* gn_b     = (const float*)d_in[30];
    const float* ffn_x_k  = (const float*)d_in[31];
    const float* W_ffn1   = (const float*)d_in[32];
    const float* W_ffn2   = (const float*)d_in[33];

    float* out_x = (float*)d_out;
    float* out_v = (float*)d_out + BTC;          // v_first output

    // ---- bump allocator over d_ws ----
    char*  ws  = (char*)d_ws;
    size_t off = 0;
    auto alloc = [&](size_t bytes) -> void* {
        void* p = ws + off;
        off += (bytes + 255) & ~(size_t)255;
        return p;
    };

    float* bufX   = (float*)alloc(BTC * 4);   // residual stream
    float* bufN   = (float*)alloc(BTC * 4);   // LN scratch (attn, then ffn)
    float* bufR   = (float*)alloc(BTC * 4);
    float* bufK   = (float*)alloc(BTC * 4);   // k raw -> k modified (in place)
    float* bufW   = (float*)alloc(BTC * 4);   // w-LoRA y -> decay (in place)
    float* bufA   = (float*)alloc(BTC * 4);   // a-LoRA y -> a (in place)
    float* bufG   = (float*)alloc(BTC * 4);   // gate
    float* bufNKK = (float*)alloc(BTC * 4);   // -kk
    float* bufKKA = (float*)alloc(BTC * 4);   // kk*a
    float* bufO   = (float*)alloc(BTC * 4);   // wkv output
    float* bufD   = (float*)alloc(BTC * 4);   // GEMM delta (attn, then ffn)

    bf16* mix = (bf16*)alloc(6 * BTC * 2);    // xr,xw,xk,xv,xa,xg
    bf16* mxr = mix + 0 * BTC;
    bf16* mxw = mix + 1 * BTC;
    bf16* mxk = mix + 2 * BTC;
    bf16* mxv = mix + 3 * BTC;
    bf16* mxa = mix + 4 * BTC;
    bf16* mxg = mix + 5 * BTC;
    bf16* h1    = mxr;                        // reuse slots 0..3 for FFN hidden (BT x 4C)
    bf16* kf    = mxa;                        // reuse slot 4 for FFN mixed input
    bf16* gated = mxg;                        // reuse slot 5 for gated attn output

    bf16* hlora = (bf16*)alloc((size_t)BT * DGd * 2);  // LoRA hidden (max BTx128)

    bf16* wrb = (bf16*)alloc((size_t)Cc * Cc * 2);
    bf16* wkb = (bf16*)alloc((size_t)Cc * Cc * 2);
    bf16* wvb = (bf16*)alloc((size_t)Cc * Cc * 2);
    bf16* wob = (bf16*)alloc((size_t)Cc * Cc * 2);
    bf16* w1b = (bf16*)alloc((size_t)Cc * DWd * 2);
    bf16* w2b = (bf16*)alloc((size_t)DWd * Cc * 2);
    bf16* a1b = (bf16*)alloc((size_t)Cc * DAd * 2);
    bf16* a2b = (bf16*)alloc((size_t)DAd * Cc * 2);
    bf16* g1b = (bf16*)alloc((size_t)Cc * DGd * 2);
    bf16* g2b = (bf16*)alloc((size_t)DGd * Cc * 2);
    bf16* f1b = (bf16*)alloc((size_t)Cc * C4 * 2);
    bf16* f2b = (bf16*)alloc((size_t)C4 * Cc * 2);

    const dim3 eb(256);
    const int  EG = 2048;   // grid-stride elementwise grid

    // ---- weight conversions (f32 -> bf16) ----
    cvt_bf16_kernel<<<EG, eb, 0, stream>>>(W_r,    wrb, (size_t)Cc * Cc);
    cvt_bf16_kernel<<<EG, eb, 0, stream>>>(W_k,    wkb, (size_t)Cc * Cc);
    cvt_bf16_kernel<<<EG, eb, 0, stream>>>(W_v,    wvb, (size_t)Cc * Cc);
    cvt_bf16_kernel<<<EG, eb, 0, stream>>>(W_o,    wob, (size_t)Cc * Cc);
    cvt_bf16_kernel<<<EG, eb, 0, stream>>>(w1,     w1b, (size_t)Cc * DWd);
    cvt_bf16_kernel<<<EG, eb, 0, stream>>>(w2,     w2b, (size_t)DWd * Cc);
    cvt_bf16_kernel<<<EG, eb, 0, stream>>>(a1,     a1b, (size_t)Cc * DAd);
    cvt_bf16_kernel<<<EG, eb, 0, stream>>>(a2,     a2b, (size_t)DAd * Cc);
    cvt_bf16_kernel<<<EG, eb, 0, stream>>>(g1,     g1b, (size_t)Cc * DGd);
    cvt_bf16_kernel<<<EG, eb, 0, stream>>>(g2,     g2b, (size_t)DGd * Cc);
    cvt_bf16_kernel<<<EG, eb, 0, stream>>>(W_ffn1, f1b, (size_t)Cc * C4);
    cvt_bf16_kernel<<<EG, eb, 0, stream>>>(W_ffn2, f2b, (size_t)C4 * Cc);

    // ---- pre-norm + attention LN + time-shift mixes ----
    ln_kernel<<<BT, eb, 0, stream>>>(x,    ln_pre_w, ln_pre_b, bufX, 1e-5f);
    ln_kernel<<<BT, eb, 0, stream>>>(bufX, ln_at_w,  ln_at_b,  bufN, 1e-5f);
    shift_mix6_kernel<<<EG, eb, 0, stream>>>(bufN, x_r, x_w, x_k, x_v, x_a, x_g,
                                             mxr, mxw, mxk, mxv, mxa, mxg);

    // ---- projections (WMMA bf16) ----
    launch_gemm<EP_F32>(mxr, wrb, bufR,  BT, Cc, Cc, stream);    // r
    launch_gemm<EP_F32>(mxk, wkb, bufK,  BT, Cc, Cc, stream);    // k (raw)
    launch_gemm<EP_F32>(mxv, wvb, out_v, BT, Cc, Cc, stream);    // v == v_first

    // ---- LoRA paths ----
    launch_gemm<EP_TANH_BF16>(mxw, w1b, hlora, BT, Cc, DWd, stream);
    launch_gemm<EP_F32>(hlora, w2b, bufW, BT, DWd, Cc, stream);
    decay_kernel<<<EG, eb, 0, stream>>>(bufW, w0);               // bufW = decay

    launch_gemm<EP_BF16>(mxa, a1b, hlora, BT, Cc, DAd, stream);
    launch_gemm<EP_F32>(hlora, a2b, bufA, BT, DAd, Cc, stream);
    sigmoid_bias_kernel<<<EG, eb, 0, stream>>>(bufA, a0);        // bufA = a

    launch_gemm<EP_SIG_BF16>(mxg, g1b, hlora, BT, Cc, DGd, stream);
    launch_gemm<EP_F32>(hlora, g2b, bufG, BT, DGd, Cc, stream);  // bufG = g

    // ---- wkv prep + scan + postmix ----
    prep_wkv_kernel<<<BT * Hc, 64, 0, stream>>>(bufK, bufA, k_k, k_a, bufNKK, bufKKA);
    wkv7_kernel<<<Bc * Hc, 64, 0, stream>>>(bufR, bufW, bufK, out_v,
                                            bufNKK, bufKKA, bufO);
    postmix_kernel<<<BT * Hc, 64, 0, stream>>>(bufO, bufR, bufK, out_v, r_k,
                                               gn_w, gn_b, bufG, gated);

    // ---- output projection + residual ----
    launch_gemm<EP_F32>(gated, wob, bufD, BT, Cc, Cc, stream);
    add_kernel<<<EG, eb, 0, stream>>>(bufX, bufD, bufX);

    // ---- channel mixing (FFN) ----
    ln_kernel<<<BT, eb, 0, stream>>>(bufX, ln_ff_w, ln_ff_b, bufN, 1e-5f);
    ffn_mix_kernel<<<EG, eb, 0, stream>>>(bufN, ffn_x_k, kf);
    launch_gemm<EP_RELU2_BF16>(kf, f1b, h1, BT, Cc, C4, stream);
    launch_gemm<EP_F32>(h1, f2b, bufD, BT, C4, Cc, stream);
    add_kernel<<<EG, eb, 0, stream>>>(bufX, bufD, out_x);
}
